// MoELayer_57449482551436
// MI455X (gfx1250) — matrix operand order repaired and checked by reading, not compiled, not run on hardware
//
#include <hip/hip_runtime.h>

// ---------------------------------------------------------------------------
// MoE layer for MI455X (gfx1250): top-2 routed grouped GEMM via bf16 WMMA,
// async global->LDS staging, L2-resident bf16 weights.
// N=4096 tokens, D=1024, E=8 experts, F=2048 hidden.
// ---------------------------------------------------------------------------

#define N_TOK 4096
#define DIM   1024
#define NEXP  8
#define FDIM  2048

typedef __attribute__((ext_vector_type(4)))  __bf16 v4bf;
typedef __attribute__((ext_vector_type(8)))  __bf16 v8bf;
typedef __attribute__((ext_vector_type(16))) __bf16 v16bf;
typedef __attribute__((ext_vector_type(8)))  float  v8f;

// LDS tile geometry (padded strides -> conflict-free ds_load_b128:
// stride_dwords % 64 == 4, so 16 lanes reading 16 different rows hit
// 16 disjoint 4-bank groups).
#define TS        32      // tokens per tile
#define XS_STRIDE 1032    // 1024 + 8 bf16 pad
#define HS_STRIDE 2056    // 2048 + 8 bf16 pad
#define SMEM_BYTES (256 + TS*XS_STRIDE*2 + TS*HS_STRIDE*2)  // 197888

// ---- fragment loaders -----------------------------------------------------
// A-frag (16x32 bf16, M rows x K): lane L<16 holds row L, K {k..k+7, k+16..k+23};
// lane L>=16 holds row L-16, K {k+8..k+15, k+24..k+31}. Caller passes p already
// offset by (k + koff); we load p[0..7] and p[16..23].
__device__ __forceinline__ v16bf ld_afrag(const __bf16* p) {
    v8bf lo = *(const v8bf*)p;
    v8bf hi = *(const v8bf*)(p + 16);
    return __builtin_shufflevector(lo, hi, 0,1,2,3,4,5,6,7,8,9,10,11,12,13,14,15);
}
// B-frag (32x16 bf16, K x N): lane L holds column N=L%16, K rows p[0..7] and
// p[16..23] (strided by ld) of the row-major [K,N] matrix; caller offsets by koff.
__device__ __forceinline__ v16bf ld_bfrag(const __bf16* p, int ld) {
    v16bf r;
#pragma unroll
    for (int i = 0; i < 8; ++i) {
        r[i]     = p[(size_t)i * ld];
        r[8 + i] = p[(size_t)(16 + i) * ld];
    }
    return r;
}
__device__ __forceinline__ v8f wmma_bf16(v16bf a, v16bf b, v8f c) {
    return __builtin_amdgcn_wmma_f32_16x16x32_bf16(false, a, false, b, (short)0, c,
                                                   false, false);
}

// Async global -> LDS copy of 16 bytes per lane (tracked by ASYNCcnt).
// dst_lds = LDS byte address (generic-pointer low 32 bits), src = global addr.
__device__ __forceinline__ void async_copy_b128(unsigned dst_lds, const void* src) {
    asm volatile("global_load_async_to_lds_b128 %0, %1, off"
                 :: "v"(dst_lds), "v"((unsigned long long)(uintptr_t)src)
                 : "memory");
}
__device__ __forceinline__ void wait_async0() {
    asm volatile("s_wait_asynccnt 0" ::: "memory");
}

// ---- fp32 -> bf16 bulk convert --------------------------------------------
__global__ void cvt_bf16_kernel(const float* __restrict__ src,
                                __bf16* __restrict__ dst, int n4) {
    int i = blockIdx.x * blockDim.x + threadIdx.x;
    if (i < n4) {
        float4 v = ((const float4*)src)[i];
        v4bf o = { (__bf16)v.x, (__bf16)v.y, (__bf16)v.z, (__bf16)v.w };
        ((v4bf*)dst)[i] = o;
    }
}

// ---- gating: softmax over 8 experts, stable top-2, 2-way softmax ----------
__global__ void gate_kernel(const float* __restrict__ x,
                            const float* __restrict__ gw,
                            const float* __restrict__ gb,
                            int*   __restrict__ topi,
                            float* __restrict__ topw) {
    int lane = threadIdx.x & 31;
    int wid  = threadIdx.x >> 5;
    int tok  = blockIdx.x * (blockDim.x >> 5) + wid;
    if (tok >= N_TOK) return;

    float acc[NEXP];
#pragma unroll
    for (int e = 0; e < NEXP; ++e) acc[e] = 0.0f;

    const float* xr = x + (size_t)tok * DIM;
    for (int d = lane; d < DIM; d += 32) {
        float xv = xr[d];
        const float4* g = (const float4*)(gw + (size_t)d * NEXP);
        float4 g0 = g[0], g1 = g[1];
        acc[0] += xv * g0.x; acc[1] += xv * g0.y;
        acc[2] += xv * g0.z; acc[3] += xv * g0.w;
        acc[4] += xv * g1.x; acc[5] += xv * g1.y;
        acc[6] += xv * g1.z; acc[7] += xv * g1.w;
    }
#pragma unroll
    for (int off = 16; off >= 1; off >>= 1)
#pragma unroll
        for (int e = 0; e < NEXP; ++e)
            acc[e] += __shfl_down(acc[e], off, 32);

    if (lane == 0) {
        float l[NEXP];
        float m = -1e30f;
#pragma unroll
        for (int e = 0; e < NEXP; ++e) { l[e] = acc[e] + gb[e]; m = fmaxf(m, l[e]); }
        float s = 0.0f;
#pragma unroll
        for (int e = 0; e < NEXP; ++e) { l[e] = __expf(l[e] - m); s += l[e]; }
        float inv = 1.0f / s;
#pragma unroll
        for (int e = 0; e < NEXP; ++e) l[e] *= inv;   // gate scores

        int i0 = 0; float v0 = l[0];
#pragma unroll
        for (int e = 1; e < NEXP; ++e) if (l[e] > v0) { v0 = l[e]; i0 = e; }
        int i1 = -1; float v1 = -1e30f;
#pragma unroll
        for (int e = 0; e < NEXP; ++e)
            if (e != i0 && l[e] > v1) { v1 = l[e]; i1 = e; }

        float w0 = 1.0f / (1.0f + __expf(v1 - v0));   // softmax over {v0,v1}
        float w1 = 1.0f - w0;
        topi[2*tok]   = i0;  topi[2*tok+1] = i1;
        topw[2*tok]   = w0;  topw[2*tok+1] = w1;
    }
}

// ---- deterministic per-expert token compaction (one wave per expert) ------
__global__ void build_lists_kernel(const int* __restrict__ topi,
                                   const float* __restrict__ topw,
                                   int*   __restrict__ tokens,
                                   float* __restrict__ wts,
                                   int*   __restrict__ counts) {
    int e = blockIdx.x;
    int lane = threadIdx.x & 31;
    int count = 0;
    for (int base = 0; base < N_TOK; base += 32) {
        int n = base + lane;
        int a = topi[2*n], b = topi[2*n+1];
        bool f0 = (a == e), f1 = (b == e);
        bool f = f0 || f1;
        float w = f0 ? topw[2*n] : topw[2*n+1];
        unsigned mask = (unsigned)__ballot(f);
        int pre = __popc(mask & ((1u << lane) - 1u));
        if (f) {
            tokens[(size_t)e * N_TOK + count + pre] = n;
            wts[(size_t)e * N_TOK + count + pre] = w;
        }
        count += __popc(mask);
    }
    if (lane == 0) counts[e] = count;
}

// ---- grouped expert MLP: per-(expert, 32-token tile) workgroup ------------
// Phase 1: H = relu(X * W1_e + b1)  (H kept in LDS as bf16)
// Phase 2: out += wt * (H * W2_e + b2)  (atomic f32 combine)
__global__ void __launch_bounds__(256)
moe_expert_kernel(const __bf16* __restrict__ xbf,
                  const __bf16* __restrict__ w1bf,
                  const __bf16* __restrict__ w2bf,
                  const float*  __restrict__ b1,
                  const float*  __restrict__ b2,
                  const int*    __restrict__ tokens,
                  const float*  __restrict__ wts,
                  const int*    __restrict__ counts,
                  float* __restrict__ out) {
    extern __shared__ char smem[];
    int*    tok = (int*)smem;                    // [32]
    float*  twt = (float*)(smem + 128);          // [32]
    __bf16* Xs  = (__bf16*)(smem + 256);                       // 32 x XS_STRIDE
    __bf16* Hs  = (__bf16*)(smem + 256 + TS * XS_STRIDE * 2);  // 32 x HS_STRIDE

    const int e    = blockIdx.y;
    const int cnt  = counts[e];
    const int row0 = blockIdx.x * TS;
    if (row0 >= cnt) return;
    const int m = min(TS, cnt - row0);

    const int tid  = threadIdx.x;
    const int lane = tid & 31;
    const int wid  = tid >> 5;
    const int col  = lane & 15;          // N index within a 16-wide subtile
    const int koff = (lane >> 4) * 8;    // A-frag K sub-offset
    const int rbase = (lane >> 4) * 8;   // C-frag row base within 16-row tile

    if (tid < TS) {
        int r = tid;
        int src = (r < m) ? (row0 + r) : row0;   // safe valid token for pads
        tok[r] = tokens[(size_t)e * N_TOK + src];
        twt[r] = (r < m) ? wts[(size_t)e * N_TOK + row0 + r] : 0.0f;
    }
    __syncthreads();

    // Stage gathered X rows into LDS via async copies (no VGPR round-trip;
    // tracked with ASYNCcnt). 16B per lane per issue, rows gathered by token.
    {
        const unsigned xs_lds = (unsigned)(uintptr_t)Xs;   // LDS byte address
        for (int idx = tid; idx < TS * (DIM / 8); idx += 256) {
            int r  = idx >> 7;             // DIM/8 = 128 chunks per row
            int c8 = (idx & 127) * 8;
            unsigned dst = xs_lds + (unsigned)(r * XS_STRIDE + c8) * 2u;
            async_copy_b128(dst, xbf + (size_t)tok[r] * DIM + c8);
        }
        wait_async0();
    }
    __syncthreads();

    const __bf16* w1e = w1bf + (size_t)e * DIM * FDIM;
    const __bf16* w2e = w2bf + (size_t)e * FDIM * DIM;

    // -------- Phase 1: H = relu(X * W1 + b1), 128 subtiles over F ----------
    for (int s = 0; s < 16; ++s) {
        const int fcol = (s * 8 + wid) * 16;
        v8f acc0 = {}, acc1 = {};
        const __bf16* a0p = Xs + (size_t)col * XS_STRIDE + koff;
        const __bf16* a1p = Xs + (size_t)(16 + col) * XS_STRIDE + koff;
        const __bf16* bp  = w1e + (size_t)koff * FDIM + fcol + col;
        // prefetch next subtile's weight strip into cache
        if (s < 15) __builtin_prefetch(bp + 128, 0, 1);
#pragma unroll 4
        for (int k = 0; k < DIM; k += 32) {
            v16bf a0 = ld_afrag(a0p + k);
            v16bf a1 = ld_afrag(a1p + k);
            v16bf b  = ld_bfrag(bp + (size_t)k * FDIM, FDIM);
            acc0 = wmma_bf16(a0, b, acc0);
            acc1 = wmma_bf16(a1, b, acc1);
        }
        const float bias = b1[(size_t)e * FDIM + fcol + col];
        __bf16* h0 = Hs + fcol + col;
#pragma unroll
        for (int r = 0; r < 8; ++r) {
            float v = fmaxf(acc0[r] + bias, 0.0f);
            h0[(size_t)(rbase + r) * HS_STRIDE] = (__bf16)v;
            float u = fmaxf(acc1[r] + bias, 0.0f);
            h0[(size_t)(16 + rbase + r) * HS_STRIDE] = (__bf16)u;
        }
    }
    __syncthreads();

    // -------- Phase 2: out += wt * (H * W2 + b2), 64 subtiles over D -------
    for (int s = 0; s < 8; ++s) {
        const int dcol = (s * 8 + wid) * 16;
        v8f acc0 = {}, acc1 = {};
        const __bf16* a0p = Hs + (size_t)col * HS_STRIDE + koff;
        const __bf16* a1p = Hs + (size_t)(16 + col) * HS_STRIDE + koff;
        const __bf16* bp  = w2e + (size_t)koff * DIM + dcol + col;
        if (s < 7) __builtin_prefetch(bp + 128, 0, 1);
#pragma unroll 4
        for (int k = 0; k < FDIM; k += 32) {
            v16bf a0 = ld_afrag(a0p + k);
            v16bf a1 = ld_afrag(a1p + k);
            v16bf b  = ld_bfrag(bp + (size_t)k * DIM, DIM);
            acc0 = wmma_bf16(a0, b, acc0);
            acc1 = wmma_bf16(a1, b, acc1);
        }
        const int d = dcol + col;
        const float bias = b2[(size_t)e * DIM + d];
#pragma unroll
        for (int r = 0; r < 8; ++r) {
            int row = rbase + r;
            if (row < m)
                atomicAdd(&out[(size_t)tok[row] * DIM + d],
                          (acc0[r] + bias) * twt[row]);
            int row2 = 16 + rbase + r;
            if (row2 < m)
                atomicAdd(&out[(size_t)tok[row2] * DIM + d],
                          (acc1[r] + bias) * twt[row2]);
        }
    }
}

// ---------------------------------------------------------------------------
extern "C" void kernel_launch(void* const* d_in, const int* in_sizes, int n_in,
                              void* d_out, int out_size, void* d_ws, size_t ws_size,
                              hipStream_t stream) {
    const float* x  = (const float*)d_in[0];
    const float* gw = (const float*)d_in[1];
    const float* gb = (const float*)d_in[2];
    const float* w1 = (const float*)d_in[3];
    const float* b1 = (const float*)d_in[4];
    const float* w2 = (const float*)d_in[5];
    const float* b2 = (const float*)d_in[6];
    float* out = (float*)d_out;
    char* ws = (char*)d_ws;

    // workspace layout (256B aligned)
    const size_t XBF  = 0;
    const size_t W1BF = XBF  + (size_t)N_TOK * DIM * 2;        //  8 MB
    const size_t W2BF = W1BF + (size_t)NEXP * DIM * FDIM * 2;  // 32 MB
    const size_t TOPI = W2BF + (size_t)NEXP * FDIM * DIM * 2;  // 32 MB
    const size_t TOPW = TOPI + (size_t)N_TOK * 2 * 4;
    const size_t TOKL = TOPW + (size_t)N_TOK * 2 * 4;
    const size_t WTSL = TOKL + (size_t)NEXP * N_TOK * 4;
    const size_t CNTS = WTSL + (size_t)NEXP * N_TOK * 4;

    __bf16* xbf  = (__bf16*)(ws + XBF);
    __bf16* w1bf = (__bf16*)(ws + W1BF);
    __bf16* w2bf = (__bf16*)(ws + W2BF);
    int*    topi = (int*)  (ws + TOPI);
    float*  topw = (float*)(ws + TOPW);
    int*    tokl = (int*)  (ws + TOKL);
    float*  wtsl = (float*)(ws + WTSL);
    int*    cnts = (int*)  (ws + CNTS);

    hipMemsetAsync(out, 0, (size_t)out_size * sizeof(float), stream);

    // fp32 -> bf16 pre-conversion (read-once)
    {
        int n4x = N_TOK * DIM / 4;                 // 1M float4
        int n4w = NEXP * DIM * FDIM / 4;           // 4M float4
        cvt_bf16_kernel<<<(n4x + 255) / 256, 256, 0, stream>>>(x,  xbf,  n4x);
        cvt_bf16_kernel<<<(n4w + 255) / 256, 256, 0, stream>>>(w1, w1bf, n4w);
        cvt_bf16_kernel<<<(n4w + 255) / 256, 256, 0, stream>>>(w2, w2bf, n4w);
    }

    // gating: 8 waves per block, one wave per token
    gate_kernel<<<N_TOK / 8, 256, 0, stream>>>(x, gw, gb, topi, topw);

    // deterministic routing lists
    build_lists_kernel<<<NEXP, 32, 0, stream>>>(topi, topw, tokl, wtsl, cnts);

    // grouped expert GEMM: (max tiles per expert) x experts
    dim3 grid(N_TOK / TS, NEXP);
    moe_expert_kernel<<<grid, 256, SMEM_BYTES, stream>>>(
        xbf, w1bf, w2bf, b1, b2, tokl, wtsl, cnts, out);
}
